// NodeAwareTokenEmbedder_31129922962205
// MI455X (gfx1250) — matrix operand order repaired and checked by reading, not compiled, not run on hardware
//
#include <hip/hip_runtime.h>

// ---------------------------------------------------------------------------
// NodeAwareTokenEmbedder for MI455X (gfx1250, wave32)
//
//   out[b,t,f] = embed_tok[tokens[b,t], f] + pos_emb[t, f]
//              + sum_n [start[b,n] <= t <= end[b,n]] * embed_node[n, f]
//
// Span term = binary-A GEMM on v_wmma_f32_16x16x32_bf16 with a two-plane
// bf16 split of embed_node (hi + lo), f32 accumulate -> ~fp32 accuracy.
// Blocking: WG = 16 waves = 16 t-tiles x 2 f-tiles. B operands staged in LDS
// once per WG via global_load_async_to_lds_b128 (ASYNCcnt), A streamed from a
// pre-swizzled workspace, token-gather rows prefetched ahead of the WMMAs.
// Inner loop software-pipelined one k-group ahead so ds_load latency is
// covered and WMMAs issue back-to-back.
// ---------------------------------------------------------------------------

typedef __attribute__((ext_vector_type(16))) __bf16 v16bf;
typedef __attribute__((ext_vector_type(8)))  float  v8f;

#define NB     4
#define TOK    1024
#define NNODES 512
#define FEAT   512

// ---- prep B: swizzle embed_node (fp32) into bf16 hi/lo WMMA B-operand layout
// 16-bit B 32x16 (KxN), this code's k order (same as A): lane L holds N = L&15, K = 16*(L>>4) + e
__global__ __launch_bounds__(256) void prep_b_kernel(
    const float* __restrict__ embed_node,
    __bf16* __restrict__ bhi, __bf16* __restrict__ blo) {
  int i = blockIdx.x * 256 + threadIdx.x;   // over NNODES*FEAT = 262144
  int k = i >> 9;                           // node index = K dim
  int f = i & 511;                          // feature  = N dim
  float x = embed_node[i];
  __bf16 h = (__bf16)x;
  __bf16 l = (__bf16)(x - (float)h);
  int ftile = f >> 4;
  int ncol  = f & 15;
  int g     = k >> 5;                       // k-group of 32
  int kl    = k & 31;
  int e     = kl & 15;                      // element within v16bf
  int half  = kl >> 4;
  int lane  = half * 16 + ncol;
  size_t off = ((size_t)((ftile * 16 + g) * 32 + lane)) * 16 + (size_t)e;
  bhi[off] = h;
  blo[off] = l;
}

// ---- prep A: build binary span mask in bf16 WMMA A-operand layout
// 16-bit A 16x32 (MxK), this code's k order (same as B): lane L holds M = L&15,
// element e holds K = e + 16*(L>>4)
__global__ __launch_bounds__(256) void prep_a_kernel(
    const int* __restrict__ starts, const int* __restrict__ ends,
    __bf16* __restrict__ aw) {
  int i = blockIdx.x * 256 + threadIdx.x;   // 4*64*16*32 = 131072 chunks
  int lane = i & 31;
  int g    = (i >> 5) & 15;                 // k-group
  int tt   = (i >> 9) & 63;                 // t tile
  int b    = i >> 15;                       // batch
  int half = lane >> 4;
  int trow = tt * 16 + (lane & 15);
  const int* sb = starts + b * NNODES;
  const int* eb = ends   + b * NNODES;
  v16bf v;
#pragma unroll
  for (int e = 0; e < 16; ++e) {
    int kl = e + (half << 4);
    int n  = g * 32 + kl;
    int s  = sb[n];
    int en = eb[n];
    v[e] = (s <= trow && trow <= en) ? (__bf16)1.0f : (__bf16)0.0f;
  }
  *(v16bf*)(aw + (size_t)i * 16) = v;
}

// ---- main: WG = 512 threads = 16 waves; wave w -> t-tile (ttblk*16+w),
// f-tiles {ftblk*2, ftblk*2+1}. B (hi/lo x 2 ft = 64 KB) async-staged in LDS.
__global__ __launch_bounds__(512) void embed_wmma_kernel(
    const __bf16* __restrict__ aw,
    const __bf16* __restrict__ bhi, const __bf16* __restrict__ blo,
    const int* __restrict__ tokens,
    const float* __restrict__ embed_tok, const float* __restrict__ pos_emb,
    float* __restrict__ out) {
  // LDS: [region = plane*2 + ftl][g][lane][e]; region: 0=hi/ft0 1=hi/ft1
  //                                                    2=lo/ft0 3=lo/ft1
  __shared__ __align__(32) __bf16 sB[4][16][32][16];   // 64 KB

  int bid   = blockIdx.x;            // 256 blocks
  int ftblk = bid & 15;              // 16 f-blocks of 2 f-tiles
  int ttblk = (bid >> 4) & 3;        // 4 t-blocks of 16 t-tiles
  int b     = bid >> 6;              // batch
  int wave  = threadIdx.x >> 5;      // 0..15
  int lane  = threadIdx.x & 31;
  int tt    = ttblk * 16 + wave;
  int ft0   = ftblk * 2;

  // ---- stage B into LDS with CDNA5 async copies (16 B per lane per issue)
  {
    const __bf16* srcs[4] = {
        bhi + (size_t)ft0 * 8192, bhi + (size_t)(ft0 + 1) * 8192,
        blo + (size_t)ft0 * 8192, blo + (size_t)(ft0 + 1) * 8192};
    unsigned ldsBase = (unsigned)(uintptr_t)&sB[0][0][0][0];
#pragma unroll
    for (int r = 0; r < 8; ++r) {
      int region = r >> 1;                               // compile-time
      int within = ((r & 1) << 9) + threadIdx.x;         // 16B chunk in region
      unsigned long long gaddr =
          (unsigned long long)(uintptr_t)(srcs[region] + (size_t)within * 8);
      unsigned laddr = ldsBase + (unsigned)(region * 16384 + within * 16);
      asm volatile("global_load_async_to_lds_b128 %0, %1, off"
                   :: "v"(laddr), "v"(gaddr) : "memory");
    }
  }

  // ---- prefetch token-embedding gather rows while the DMA + WMMAs run
  int half = lane >> 4;
  int n    = lane & 15;
  int f0   = ft0 * 16 + n;
  int tokv[8];
#pragma unroll
  for (int r = 0; r < 8; ++r) {
    int t = tt * 16 + r + 8 * half;
    tokv[r] = tokens[b * TOK + t];
  }
#pragma unroll
  for (int r = 0; r < 8; ++r) {
    __builtin_prefetch(&embed_tok[(size_t)tokv[r] * FEAT + f0], 0, 1);
  }

  asm volatile("s_wait_asynccnt 0x0" ::: "memory");
  __syncthreads();

  // ---- 16 k-groups x (2 f-tiles x 2 planes) WMMAs, software-pipelined one
  // group ahead so ds_load/global_load latency never blocks WMMA issue.
  const v16bf* Ap  = (const v16bf*)aw + (size_t)(b * 64 + tt) * 16 * 32 + lane;
  const v16bf* sBv = (const v16bf*)&sB[0][0][0][0];
  v8f c0 = {};  // f-tile ft0 accumulator
  v8f c1 = {};  // f-tile ft0+1 accumulator (independent chain)

  v16bf a   = Ap[0];
  v16bf bh0 = sBv[(0 * 16) * 32 + lane];
  v16bf bh1 = sBv[(1 * 16) * 32 + lane];
  v16bf bl0 = sBv[(2 * 16) * 32 + lane];
  v16bf bl1 = sBv[(3 * 16) * 32 + lane];
#pragma unroll
  for (int g = 0; g < 16; ++g) {
    v16bf an, nh0, nh1, nl0, nl1;
    if (g < 15) {                       // static under full unroll
      an  = Ap[(g + 1) * 32];
      nh0 = sBv[(0 * 16 + g + 1) * 32 + lane];
      nh1 = sBv[(1 * 16 + g + 1) * 32 + lane];
      nl0 = sBv[(2 * 16 + g + 1) * 32 + lane];
      nl1 = sBv[(3 * 16 + g + 1) * 32 + lane];
    }
    c0 = __builtin_amdgcn_wmma_f32_16x16x32_bf16(false, a, false, bh0,
                                                 (short)0, c0, false, false);
    c1 = __builtin_amdgcn_wmma_f32_16x16x32_bf16(false, a, false, bh1,
                                                 (short)0, c1, false, false);
    c0 = __builtin_amdgcn_wmma_f32_16x16x32_bf16(false, a, false, bl0,
                                                 (short)0, c0, false, false);
    c1 = __builtin_amdgcn_wmma_f32_16x16x32_bf16(false, a, false, bl1,
                                                 (short)0, c1, false, false);
    if (g < 15) {
      a = an; bh0 = nh0; bh1 = nh1; bl0 = nl0; bl1 = nl1;
    }
  }

  // ---- epilogue: C/D layout M = r + 8*(lane>>4), N = lane&15
  int f1 = f0 + 16;
#pragma unroll
  for (int r = 0; r < 8; ++r) {
    int t = tt * 16 + r + 8 * half;
    size_t rowTok = (size_t)tokv[r] * FEAT;
    size_t rowPos = (size_t)t * FEAT;
    size_t rowOut = (size_t)(b * TOK + t) * FEAT;
    out[rowOut + f0] = c0[r] + embed_tok[rowTok + f0] + pos_emb[rowPos + f0];
    out[rowOut + f1] = c1[r] + embed_tok[rowTok + f1] + pos_emb[rowPos + f1];
  }
}

extern "C" void kernel_launch(void* const* d_in, const int* in_sizes, int n_in,
                              void* d_out, int out_size, void* d_ws, size_t ws_size,
                              hipStream_t stream) {
  (void)in_sizes; (void)n_in; (void)out_size; (void)ws_size;
  const int*   tokens     = (const int*)d_in[0];
  const int*   starts     = (const int*)d_in[1];
  const int*   ends       = (const int*)d_in[2];
  const float* embed_tok  = (const float*)d_in[3];
  const float* embed_node = (const float*)d_in[4];
  const float* pos_emb    = (const float*)d_in[5];
  float* out = (float*)d_out;

  // Workspace layout: Bhi (512 KB) | Blo (512 KB) | A (4 MB)  -> 5 MB total
  __bf16* bhi = (__bf16*)d_ws;
  __bf16* blo = bhi + (size_t)NNODES * FEAT;
  __bf16* aw  = blo + (size_t)NNODES * FEAT;

  prep_b_kernel<<<(NNODES * FEAT) / 256, 256, 0, stream>>>(embed_node, bhi, blo);
  prep_a_kernel<<<(NB * 64 * 16 * 32) / 256, 256, 0, stream>>>(starts, ends, aw);
  embed_wmma_kernel<<<256, 512, 0, stream>>>(
      aw, bhi, blo, tokens, embed_tok, pos_emb, out);
}
